// KeypointInitializer_33406255628288
// MI455X (gfx1250) — matrix operand order repaired and checked by reading, not compiled, not run on hardware
//
#include <hip/hip_runtime.h>
#include <hip/hip_bf16.h>
#include <math.h>

// Problem constants (fixed by the reference)
#define BB 32
#define NN 4096
#define DD 256
#define KK 32
#define DK (DD*KK)          // 8192
#define NCHUNK 16           // 256 n-rows per block -> 16 chunks per batch
#define WAVES 4             // 4 wave32 per block (128 threads)
#define SUBT 4              // 4 subtiles of 16 rows per wave -> 64 rows/wave
#define HS 268              // padded LDS row stride (bf16 elems): 8B-aligned rows, conflict-free WMMA gathers
#define QS 268

typedef __attribute__((ext_vector_type(16))) __bf16        v16bf;
typedef __attribute__((ext_vector_type(8)))  float         v8f;
typedef __attribute__((ext_vector_type(8)))  unsigned int  v8u;

__device__ __forceinline__ unsigned short f2bf(float f) {
  unsigned u = __builtin_bit_cast(unsigned, f);
  u += 0x7FFFu + ((u >> 16) & 1u);          // round-to-nearest-even
  return (unsigned short)(u >> 16);
}
__device__ __forceinline__ unsigned pack2(float lo, float hi) {
  return (unsigned)f2bf(lo) | ((unsigned)f2bf(hi) << 16);
}

// ---- K0: zero the kp_scalars / kp_vecs output tail ---------------------------
__global__ void k0_zero(float* __restrict__ p, int count) {
  int i = blockIdx.x * blockDim.x + threadIdx.x;
  if (i < count) p[i] = 0.0f;
}

// ---- K1a/K1b: deterministic two-stage column mean over N ---------------------
__global__ __launch_bounds__(256)
void k1a_partial(const float* __restrict__ h, float* __restrict__ pmean) {
  int b = blockIdx.x >> 4, s = blockIdx.x & 15, t = threadIdx.x;
  const float* base = h + ((size_t)(b * NN) + (size_t)s * 256) * DD + t;
  float acc = 0.0f;
  for (int i = 0; i < 256; ++i) acc += base[(size_t)i * DD];
  pmean[(size_t)blockIdx.x * DD + t] = acc;
}
__global__ __launch_bounds__(256)
void k1b_mean(const float* __restrict__ pmean, float* __restrict__ mean) {
  int b = blockIdx.x, t = threadIdx.x;
  float acc = 0.0f;
  for (int s = 0; s < 16; ++s) acc += pmean[(size_t)(b * 16 + s) * DD + t];
  mean[b * DD + t] = acc * (1.0f / NN);
}

// ---- K2: M = W_dst^T @ W_src (256x256) ---------------------------------------
__global__ __launch_bounds__(256)
void k2_mmat(const float* __restrict__ Wd, const float* __restrict__ Ws,
             float* __restrict__ M) {
  int d = blockIdx.x, dp = threadIdx.x;
  float acc = 0.0f;
  for (int e = 0; e < DD; ++e) acc += Wd[e * DD + d] * Ws[e * DD + dp];
  M[d * DD + dp] = acc;
}

// ---- K3a: kp_lin = SiLU(mean @ W_kp^T + b_kp) --------------------------------
__global__ __launch_bounds__(256)
void k3a_kp(const float* __restrict__ mean, const float* __restrict__ Wkp,
            const float* __restrict__ bkp, float* __restrict__ kplin) {
  __shared__ float ms[DD];
  int b = blockIdx.x >> 5, jb = blockIdx.x & 31, t = threadIdx.x;
  ms[t] = mean[b * DD + t];
  __syncthreads();
  int j = jb * 256 + t;
  const float* wr = Wkp + (size_t)j * DD;
  float acc = bkp[j];
  for (int d = 0; d < DD; ++d) acc += ms[d] * wr[d];
  float sig = 1.0f / (1.0f + __expf(-acc));
  kplin[(size_t)b * DK + j] = acc * sig;
}

// ---- K3b: LayerNorm statistics per batch -------------------------------------
__global__ __launch_bounds__(256)
void k3b_stats(const float* __restrict__ kplin, float* __restrict__ stats) {
  __shared__ float ss[256], sq[256];
  int b = blockIdx.x, t = threadIdx.x;
  float s = 0.0f, q2 = 0.0f;
  for (int i = t; i < DK; i += 256) {
    float v = kplin[(size_t)b * DK + i];
    s += v; q2 += v * v;
  }
  ss[t] = s; sq[t] = q2;
  __syncthreads();
  for (int o = 128; o > 0; o >>= 1) {
    if (t < o) { ss[t] += ss[t + o]; sq[t] += sq[t + o]; }
    __syncthreads();
  }
  if (t == 0) {
    float mu = ss[0] / DK;
    float var = sq[0] / DK - mu * mu;
    stats[b * 2] = mu;
    stats[b * 2 + 1] = rsqrtf(var + 1e-5f);
  }
}

// ---- K4: q[b,k,:] = LN(kp)[b,k,:] @ M ----------------------------------------
__global__ __launch_bounds__(256)
void k4_q(const float* __restrict__ kplin, const float* __restrict__ stats,
          const float* __restrict__ gamma, const float* __restrict__ beta,
          const float* __restrict__ M, float* __restrict__ q) {
  __shared__ float kpn[DD];
  int b = blockIdx.x >> 5, k = blockIdx.x & 31, t = threadIdx.x;
  float mu = stats[b * 2], rstd = stats[b * 2 + 1];
  int j = k * DD + t;
  kpn[t] = (kplin[(size_t)b * DK + j] - mu) * rstd * gamma[j] + beta[j];
  __syncthreads();
  float acc = 0.0f;
  for (int d = 0; d < DD; ++d) acc += kpn[d] * M[d * DD + t];
  q[((size_t)(b * KK) + k) * DD + t] = acc;
}

// ---- K5: flash attention scores (bf16 WMMA) + online softmax + pos accum -----
__global__ __launch_bounds__(128)
void k5_attn(const float* __restrict__ h, const float* __restrict__ pos,
             const float* __restrict__ q, float* __restrict__ part) {
  __shared__ __align__(16) unsigned short q_s[KK * QS];        // 16.75 KB
  __shared__ __align__(16) unsigned short h_s[WAVES][16 * HS]; // 33.5 KB
  const int b = blockIdx.x >> 4;
  const int chunk = blockIdx.x & 15;
  const int tid = threadIdx.x;
  const int lane = tid & 31;
  const int w = tid >> 5;

  // Stage all 32 q-rows (bf16) once per block
  for (int i = tid; i < KK * DD; i += 128) {
    int r = i >> 8, c = i & 255;
    q_s[r * QS + c] = f2bf(q[((size_t)b * KK + r) * DD + c]);
  }
  __syncthreads();

  const int half = lane >> 4;   // lane-half selects the K-subrange of WMMA operands
  const int m16 = lane & 15;
  float mrun[2] = {-1e30f, -1e30f}, srun[2] = {0.0f, 0.0f};
  float px[2] = {0.0f, 0.0f}, py[2] = {0.0f, 0.0f}, pz[2] = {0.0f, 0.0f};
  const int n_wave = chunk * 256 + w * 64;

  for (int st = 0; st < SUBT; ++st) {
    const int n_base = n_wave + st * 16;
    // Stage 16x256 f32 h-tile -> bf16 LDS (coalesced float4 loads)
    for (int r = 0; r < 16; ++r) {
      const float* src = h + ((size_t)(b * NN + n_base + r)) * DD;
#pragma unroll
      for (int i = 0; i < 2; ++i) {
        int d = (lane + 32 * i) * 4;
        float4 v = *(const float4*)(src + d);
        uint2 pk; pk.x = pack2(v.x, v.y); pk.y = pack2(v.z, v.w);
        *(uint2*)(&h_s[w][r * HS + d]) = pk;
        if (st + 1 < SUBT) __builtin_prefetch(src + 16 * DD + d, 0, 0);
      }
    }
    __syncthreads();

    // S^T(16n x 16k) += H(16x32) * Q^T(32x16) over 8 d-slices, both k-tiles
    v8f c0 = {}, c1 = {};
#pragma unroll
    for (int j = 0; j < 8; ++j) {
      const int d0 = 32 * j;
      v8u ab;
#pragma unroll
      for (int v = 0; v < 8; ++v) {
        int off = (v < 4) ? (2 * v + 8 * half) : (16 + 2 * (v - 4) + 8 * half);
        ab[v] = *(const unsigned*)(&h_s[w][m16 * HS + d0 + off]);
      }
      v16bf A = __builtin_bit_cast(v16bf, ab);
      v8u bb0, bb1;
#pragma unroll
      for (int v = 0; v < 8; ++v) {
        int off = (v < 4) ? (2 * v + 8 * half) : (16 + 2 * (v - 4) + 8 * half);
        bb0[v] = *(const unsigned*)(&q_s[m16 * QS + d0 + off]);
        bb1[v] = *(const unsigned*)(&q_s[(16 + m16) * QS + d0 + off]);
      }
      v16bf B0 = __builtin_bit_cast(v16bf, bb0);
      v16bf B1 = __builtin_bit_cast(v16bf, bb1);
      c0 = __builtin_amdgcn_wmma_f32_16x16x32_bf16(false, A, false, B0, (short)0, c0, false, false);
      c1 = __builtin_amdgcn_wmma_f32_16x16x32_bf16(false, A, false, B1, (short)0, c1, false, false);
    }

    // Online softmax: lane holds k=(t*16 + lane%16), rows n = n_base + r + 8*half
    v8f cc[2] = {c0, c1};
#pragma unroll
    for (int t = 0; t < 2; ++t) {
      float mt = -1e30f;
#pragma unroll
      for (int r = 0; r < 8; ++r) mt = fmaxf(mt, cc[t][r] * 0.0625f);
      float mnew = fmaxf(mrun[t], mt);
      float sc = __expf(mrun[t] - mnew);
      srun[t] *= sc; px[t] *= sc; py[t] *= sc; pz[t] *= sc;
#pragma unroll
      for (int r = 0; r < 8; ++r) {
        float e = __expf(cc[t][r] * 0.0625f - mnew);
        int n = n_base + r + 8 * half;
        const float* pp = pos + (size_t)(b * NN + n) * 3;
        srun[t] += e;
        px[t] += e * pp[0]; py[t] += e * pp[1]; pz[t] += e * pp[2];
      }
      mrun[t] = mnew;
    }
    __syncthreads();
  }

  // Combine the two lane-halves (same k, disjoint n) and emit per-wave partials
#pragma unroll
  for (int t = 0; t < 2; ++t) {
    float om = __shfl_xor(mrun[t], 16, 32);
    float os = __shfl_xor(srun[t], 16, 32);
    float ox = __shfl_xor(px[t], 16, 32);
    float oy = __shfl_xor(py[t], 16, 32);
    float oz = __shfl_xor(pz[t], 16, 32);
    float mm = fmaxf(mrun[t], om);
    float e1 = __expf(mrun[t] - mm), e2 = __expf(om - mm);
    float S = srun[t] * e1 + os * e2;
    float X = px[t] * e1 + ox * e2;
    float Y = py[t] * e1 + oy * e2;
    float Z = pz[t] * e1 + oz * e2;
    if (lane < 16) {
      int wc = chunk * WAVES + w;                       // 0..63
      size_t idx = (((size_t)b * 64 + wc) * KK + t * 16 + lane) * 5;
      part[idx] = mm; part[idx + 1] = S; part[idx + 2] = X;
      part[idx + 3] = Y; part[idx + 4] = Z;
    }
  }
}

// ---- K6: deterministic merge of 64 partials per (b,k) -> kp_pos --------------
__global__ void k6_finalize(const float* __restrict__ part, float* __restrict__ out) {
  int b = blockIdx.x, k = threadIdx.x;   // 32 threads
  float m = -1e30f, s = 0.0f, x = 0.0f, y = 0.0f, z = 0.0f;
  for (int c = 0; c < 64; ++c) {
    size_t idx = (((size_t)b * 64 + c) * KK + k) * 5;
    float pm = part[idx], ps = part[idx + 1];
    float pX = part[idx + 2], pY = part[idx + 3], pZ = part[idx + 4];
    float mm = fmaxf(m, pm);
    float e1 = __expf(m - mm), e2 = __expf(pm - mm);
    s = s * e1 + ps * e2;
    x = x * e1 + pX * e2;
    y = y * e1 + pY * e2;
    z = z * e1 + pZ * e2;
    m = mm;
  }
  float inv = 1.0f / s;
  out[(b * KK + k) * 3 + 0] = x * inv;
  out[(b * KK + k) * 3 + 1] = y * inv;
  out[(b * KK + k) * 3 + 2] = z * inv;
}

extern "C" void kernel_launch(void* const* d_in, const int* in_sizes, int n_in,
                              void* d_out, int out_size, void* d_ws, size_t ws_size,
                              hipStream_t stream) {
  const float* h     = (const float*)d_in[0];
  const float* pos   = (const float*)d_in[1];
  const float* Wsrc  = (const float*)d_in[2];
  const float* Wdst  = (const float*)d_in[3];
  const float* Wkp   = (const float*)d_in[4];
  const float* bkp   = (const float*)d_in[5];
  const float* gamma = (const float*)d_in[6];
  const float* beta  = (const float*)d_in[7];
  float* out = (float*)d_out;
  float* ws  = (float*)d_ws;

  // Workspace layout (floats); total ~4.03 MB
  float* mean  = ws;                       // B*D            = 8192
  float* pmean = mean  + BB * DD;          // B*16*D         = 131072
  float* Mmat  = pmean + BB * 16 * DD;     // D*D            = 65536
  float* kplin = Mmat  + DD * DD;          // B*D*K          = 262144
  float* stats = kplin + (size_t)BB * DK;  // 2*B            = 64
  float* qbuf  = stats + 2 * BB;           // B*K*D          = 262144
  float* part  = qbuf  + (size_t)BB * KK * DD; // B*64*K*5   = 327680

  // kp_scalars + kp_vecs outputs are zeros per the reference
  int tail = out_size - BB * KK * 3;
  k0_zero<<<(tail + 255) / 256, 256, 0, stream>>>(out + BB * KK * 3, tail);

  k1a_partial<<<BB * 16, 256, 0, stream>>>(h, pmean);
  k1b_mean  <<<BB,       256, 0, stream>>>(pmean, mean);
  k2_mmat   <<<DD,       256, 0, stream>>>(Wdst, Wsrc, Mmat);
  k3a_kp    <<<BB * 32,  256, 0, stream>>>(mean, Wkp, bkp, kplin);
  k3b_stats <<<BB,       256, 0, stream>>>(kplin, stats);
  k4_q      <<<BB * KK,  256, 0, stream>>>(kplin, stats, gamma, beta, Mmat, qbuf);
  k5_attn   <<<BB * NCHUNK, 128, 0, stream>>>(h, pos, qbuf, part);
  k6_finalize<<<BB, KK, 0, stream>>>(part, out);
}
